// ChildSumTreeLSTM_63453846831535
// MI455X (gfx1250) — compile-verified, matrix operand order checked by reading, not run on hardware
//
#include <hip/hip_runtime.h>
#include <hip/hip_bf16.h>

typedef __bf16 bf16_t;
typedef __attribute__((ext_vector_type(16))) __bf16 v16bf;
typedef __attribute__((ext_vector_type(8)))  __bf16 v8bf;
typedef __attribute__((ext_vector_type(8)))  float   v8f;

#define IN_DIM  300
#define IN_PAD  320
#define MEM     150
#define MEM_PAD 160

// gate order: 0=i, 1=f, 2=u, 3=o
#define XSZ (MEM_PAD * IN_PAD)    // 51200 elems per x-gate (transposed, padded)
#define HSZ (MEM_PAD * MEM_PAD)   // 25600 elems per h-gate

#define WMMA_BF16(ACC, AV_, BV_) \
  (ACC) = __builtin_amdgcn_wmma_f32_16x16x32_bf16(false, (AV_), false, (BV_), (short)0, (ACC), false, false)

__device__ __forceinline__ float sigmoidf_(float x) { return 1.0f / (1.0f + __expf(-x)); }

// ---------------------------------------------------------------------------
// Convert the 8 fp32 weight matrices into bf16, transposed + zero-padded:
//   x-gates:  WxT[g][n][k], n in [0,160), k in [0,320)
//   h-gates:  WhT[g][n][k], n in [0,160), k in [0,160)
// ---------------------------------------------------------------------------
__global__ void convert_weights(const float* __restrict__ Wix, const float* __restrict__ Wfx,
                                const float* __restrict__ Wux, const float* __restrict__ Wox,
                                const float* __restrict__ Wih, const float* __restrict__ Wfh,
                                const float* __restrict__ Wuh, const float* __restrict__ Woh,
                                bf16_t* __restrict__ wbT) {
  int idx = blockIdx.x * blockDim.x + threadIdx.x;
  if (idx < 4 * XSZ) {
    int g = idx / XSZ, r = idx % XSZ;
    int nc = r / IN_PAD, kk = r % IN_PAD;
    const float* W = (g == 0) ? Wix : (g == 1) ? Wfx : (g == 2) ? Wux : Wox;
    float v = (nc < MEM && kk < IN_DIM) ? W[(size_t)kk * MEM + nc] : 0.0f;
    wbT[idx] = (bf16_t)v;
  } else {
    int j = idx - 4 * XSZ;
    if (j < 4 * HSZ) {
      int g = j / HSZ, r = j % HSZ;
      int nc = r / MEM_PAD, kk = r % MEM_PAD;
      const float* W = (g == 0) ? Wih : (g == 1) ? Wfh : (g == 2) ? Wuh : Woh;
      float v = (nc < MEM && kk < MEM) ? W[(size_t)kk * MEM + nc] : 0.0f;
      wbT[4 * XSZ + j] = (bf16_t)v;
    }
  }
}

// ---------------------------------------------------------------------------
// One TreeLSTM level. Block = 16-node M-tile, 10 wave32s = 10 column tiles.
// Templated on IS_LEAF so the hot leaf level (75% of nodes) has zero
// per-chunk branching and the non-leaf path is straight-line unrolled.
// ---------------------------------------------------------------------------
template <bool IS_LEAF>
__global__ __launch_bounds__(320)
void treelstm_level(const float* __restrict__ x_base,   // [n,300] this level's embs
                    const float* __restrict__ hprev,    // [4n,150] prev level h (null at leaf)
                    const float* __restrict__ cprev,    // [4n,160] prev level c (null at leaf)
                    float* __restrict__ hout,           // [n,150] into d_out
                    float* __restrict__ cout,           // [n,160] workspace
                    const bf16_t* __restrict__ wbT,
                    const float* __restrict__ b_ix, const float* __restrict__ b_fx,
                    const float* __restrict__ b_ux, const float* __restrict__ b_ox,
                    const float* __restrict__ b_ih, const float* __restrict__ b_fh,
                    const float* __restrict__ b_uh, const float* __restrict__ b_oh,
                    int n) {
  __shared__ __align__(16) bf16_t xs[16 * IN_PAD];        // 10240 B
  __shared__ __align__(16) bf16_t hsum[16 * MEM_PAD];     //  5120 B
  __shared__ __align__(16) bf16_t hch[4 * 16 * MEM_PAD];  // 20480 B

  const int tid = threadIdx.x;
  const int node0 = blockIdx.x * 16;

  // ---- stage x tile (fp32 -> bf16, zero-padded) ----
  for (int idx = tid; idx < 16 * IN_PAD; idx += 320) {
    int m = idx / IN_PAD, k = idx % IN_PAD;
    int node = node0 + m;
    float v = (node < n && k < IN_DIM) ? x_base[(size_t)node * IN_DIM + k] : 0.0f;
    xs[idx] = (bf16_t)v;
  }
  // ---- stage children h (bf16) + fp32 child-sum ----
  if (!IS_LEAF) {
    for (int idx = tid; idx < 16 * MEM_PAD; idx += 320) {
      int m = idx / MEM_PAD, col = idx % MEM_PAD;
      int node = node0 + m;
      float s = 0.0f;
      if (node < n && col < MEM) {
#pragma unroll
        for (int k = 0; k < 4; ++k) {
          float hv = hprev[(size_t)(4 * node + k) * MEM + col];
          hch[((size_t)(k * 16 + m)) * MEM_PAD + col] = (bf16_t)hv;
          s += hv;
        }
      } else {
#pragma unroll
        for (int k = 0; k < 4; ++k)
          hch[((size_t)(k * 16 + m)) * MEM_PAD + col] = (bf16_t)0.0f;
      }
      hsum[idx] = (bf16_t)s;
    }
  }
  __syncthreads();

  const int wave = tid >> 5;      // 0..9 -> column tile
  const int lane = tid & 31;
  const int nl   = lane & 15;     // A: row M, B: col N, C/D: col N
  const int half = lane >> 4;     // K-half selector
  const int colBase = wave * 16;

  const bf16_t* WxT = wbT;
  const bf16_t* WhT = wbT + 4 * XSZ;

  v8f acc_i = {}, acc_f = {}, acc_u = {}, acc_o = {};
  v8f accc[4] = {{}, {}, {}, {}};

  union AV { v16bf v; v8bf h[2]; };

  // ---- x @ Wx for gates (f gate only exists at non-leaf levels) ----
  {
    const bf16_t* xrow = xs + (size_t)nl * IN_PAD;
    const bf16_t* wcol = WxT + (size_t)(colBase + nl) * IN_PAD;
#pragma unroll
    for (int kk = 0; kk < IN_PAD / 32; ++kk) {
      const int base = kk * 32 + half * 8;
      AV a;
      a.h[0] = *(const v8bf*)(xrow + base);
      a.h[1] = *(const v8bf*)(xrow + base + 16);
      AV b;
      b.h[0] = *(const v8bf*)(wcol + 0 * XSZ + base);
      b.h[1] = *(const v8bf*)(wcol + 0 * XSZ + base + 16);
      WMMA_BF16(acc_i, a.v, b.v);
      if (!IS_LEAF) {
        b.h[0] = *(const v8bf*)(wcol + 1 * XSZ + base);
        b.h[1] = *(const v8bf*)(wcol + 1 * XSZ + base + 16);
        WMMA_BF16(acc_f, a.v, b.v);
      }
      b.h[0] = *(const v8bf*)(wcol + 2 * XSZ + base);
      b.h[1] = *(const v8bf*)(wcol + 2 * XSZ + base + 16);
      WMMA_BF16(acc_u, a.v, b.v);
      b.h[0] = *(const v8bf*)(wcol + 3 * XSZ + base);
      b.h[1] = *(const v8bf*)(wcol + 3 * XSZ + base + 16);
      WMMA_BF16(acc_o, a.v, b.v);
    }
  }

  // ---- h_sum @ Wh (i,u,o) and per-child h @ W_fh ----
  if (!IS_LEAF) {
    const bf16_t* hsrow = hsum + (size_t)nl * MEM_PAD;
    const bf16_t* wcol  = WhT + (size_t)(colBase + nl) * MEM_PAD;
#pragma unroll
    for (int kk = 0; kk < MEM_PAD / 32; ++kk) {
      const int base = kk * 32 + half * 8;
      AV a;
      a.h[0] = *(const v8bf*)(hsrow + base);
      a.h[1] = *(const v8bf*)(hsrow + base + 16);
      AV b;
      b.h[0] = *(const v8bf*)(wcol + 0 * HSZ + base);
      b.h[1] = *(const v8bf*)(wcol + 0 * HSZ + base + 16);
      WMMA_BF16(acc_i, a.v, b.v);
      b.h[0] = *(const v8bf*)(wcol + 2 * HSZ + base);
      b.h[1] = *(const v8bf*)(wcol + 2 * HSZ + base + 16);
      WMMA_BF16(acc_u, a.v, b.v);
      b.h[0] = *(const v8bf*)(wcol + 3 * HSZ + base);
      b.h[1] = *(const v8bf*)(wcol + 3 * HSZ + base + 16);
      WMMA_BF16(acc_o, a.v, b.v);
      AV bf;
      bf.h[0] = *(const v8bf*)(wcol + 1 * HSZ + base);
      bf.h[1] = *(const v8bf*)(wcol + 1 * HSZ + base + 16);
#pragma unroll
      for (int k = 0; k < 4; ++k) {
        const bf16_t* hcr = hch + ((size_t)(k * 16 + nl)) * MEM_PAD + base;
        AV ac;
        ac.h[0] = *(const v8bf*)(hcr);
        ac.h[1] = *(const v8bf*)(hcr + 16);
        WMMA_BF16(accc[k], ac.v, bf.v);
      }
    }
  }

  // ---- gates + recurrence (fp32), write h/c ----
  const int col = colBase + nl;
  float bi = 0.f, bff = 0.f, bu = 0.f, bo = 0.f;
  if (col < MEM) {
    bi  = b_ix[col] + b_ih[col];
    bff = b_fx[col] + b_fh[col];
    bu  = b_ux[col] + b_uh[col];
    bo  = b_ox[col] + b_oh[col];
  }
#pragma unroll
  for (int r = 0; r < 8; ++r) {
    int m = r + 8 * half;
    int node = node0 + m;
    if (node < n && col < MEM) {
      float ig = sigmoidf_(acc_i[r] + bi);
      float ug = tanhf(acc_u[r] + bu);
      float og = sigmoidf_(acc_o[r] + bo);
      float fc = 0.0f;
      if (!IS_LEAF) {
        float fx = acc_f[r] + bff;
#pragma unroll
        for (int k = 0; k < 4; ++k) {
          float fg = sigmoidf_(accc[k][r] + fx);
          fc += fg * cprev[(size_t)(4 * node + k) * MEM_PAD + col];
        }
      }
      float c = ig * ug + fc;
      float h = og * tanhf(c);
      hout[(size_t)node * MEM + col] = h;
      cout[(size_t)node * MEM_PAD + col] = c;
    }
  }
}

// ---------------------------------------------------------------------------
extern "C" void kernel_launch(void* const* d_in, const int* in_sizes, int n_in,
                              void* d_out, int out_size, void* d_ws, size_t ws_size,
                              hipStream_t stream) {
  (void)in_sizes; (void)n_in; (void)out_size; (void)ws_size;
  const float* embs = (const float*)d_in[0];
  const float* W_ix = (const float*)d_in[1];  const float* b_ix = (const float*)d_in[2];
  const float* W_fx = (const float*)d_in[3];  const float* b_fx = (const float*)d_in[4];
  const float* W_ux = (const float*)d_in[5];  const float* b_ux = (const float*)d_in[6];
  const float* W_ox = (const float*)d_in[7];  const float* b_ox = (const float*)d_in[8];
  const float* W_ih = (const float*)d_in[9];  const float* b_ih = (const float*)d_in[10];
  const float* W_fh = (const float*)d_in[11]; const float* b_fh = (const float*)d_in[12];
  const float* W_uh = (const float*)d_in[13]; const float* b_uh = (const float*)d_in[14];
  const float* W_oh = (const float*)d_in[15]; const float* b_oh = (const float*)d_in[16];
  float* out = (float*)d_out;

  bf16_t* wbT = (bf16_t*)d_ws;
  const size_t WB_ELEMS = 4 * (size_t)XSZ + 4 * (size_t)HSZ;   // 307200
  float* cbuf = (float*)((char*)d_ws + WB_ELEMS * sizeof(bf16_t)); // 614400 B, 256-aligned

  {
    int total = (int)WB_ELEMS;
    convert_weights<<<(total + 255) / 256, 256, 0, stream>>>(
        W_ix, W_fx, W_ux, W_ox, W_ih, W_fh, W_uh, W_oh, wbT);
  }

  const int sizes[9] = {65536, 16384, 4096, 1024, 256, 64, 16, 4, 1};
  long off = 0, off_prev = 0;
  for (int d = 0; d < 9; ++d) {
    int nlev = sizes[d];
    const float* x  = embs + off * IN_DIM;
    const float* hp = d ? out  + off_prev * MEM     : nullptr;
    const float* cp = d ? cbuf + off_prev * MEM_PAD : nullptr;
    float* ho = out  + off * MEM;
    float* co = cbuf + off * MEM_PAD;
    int blocks = (nlev + 15) / 16;
    if (d == 0) {
      treelstm_level<true><<<blocks, 320, 0, stream>>>(
          x, hp, cp, ho, co, wbT,
          b_ix, b_fx, b_ux, b_ox, b_ih, b_fh, b_uh, b_oh, nlev);
    } else {
      treelstm_level<false><<<blocks, 320, 0, stream>>>(
          x, hp, cp, ho, co, wbT,
          b_ix, b_fx, b_ux, b_ox, b_ih, b_fh, b_uh, b_oh, nlev);
    }
    off_prev = off;
    off += nlev;
  }
}
